// NetGCN_51788715655649
// MI455X (gfx1250) — matrix operand order repaired and checked by reading, not compiled, not run on hardware
//
#include <hip/hip_runtime.h>
#include <math.h>

// ---------------- problem constants ----------------
#define NN     60000
#define EE     960000
#define GG     500
#define F_IN   128
#define H0     256
#define H1     256
#define RR     4
#define HOM    31
#define BN_EPS 1e-5f

// ---------------- WMMA types ----------------
typedef __attribute__((ext_vector_type(16))) __bf16 v16bf;
typedef __attribute__((ext_vector_type(8)))  float  v8f;

union Frag {
    v16bf v;
    unsigned u[8];
};

__device__ __forceinline__ unsigned short bf16bits(float f) {
    unsigned u = __builtin_bit_cast(unsigned, f);
    u += 0x7FFFu + ((u >> 16) & 1u);          // round-to-nearest-even
    return (unsigned short)(u >> 16);
}
__device__ __forceinline__ unsigned pack2(float lo, float hi) {
    return (unsigned)bf16bits(lo) | ((unsigned)bf16bits(hi) << 16);
}
__device__ __forceinline__ float bf2f(unsigned short b) {
    unsigned u = (unsigned)b << 16;
    return __builtin_bit_cast(float, u);
}

// --------- CDNA5 async global->LDS copy (bypasses VGPRs, ASYNCcnt-tracked) ---------
__device__ __forceinline__ unsigned lds_addr_of(void* p) {
    return (unsigned)(unsigned long long)(__attribute__((address_space(3))) void*)p;
}
__device__ __forceinline__ void async_copy_b128(unsigned lds_byte_addr,
                                                unsigned long long gaddr) {
    asm volatile("global_load_async_to_lds_b128 %0, %1, off"
                 :: "v"(lds_byte_addr), "v"(gaddr) : "memory");
}
__device__ __forceinline__ void wait_asynccnt0() {
    asm volatile("s_wait_asynccnt 0x0" ::: "memory");
}

// ------------- Weight pre-pack: fp32 [K][256] -> WMMA B-fragment layout -------------
// out[((ntile*(K/32) + kstep)*32 + lane)*8 + j] = bf16pair(B[kstep*32 + kk .. +1][ntile*16+lane%16]), kk = j<4 ? 8*(lane/16)+2j : 16+8*(lane/16)+2(j-4)
__global__ void pack_b_kernel(const float* __restrict__ B, unsigned* __restrict__ out, int K) {
    const int idx    = blockIdx.x * 256 + threadIdx.x;
    const int ksteps = K >> 5;
    const int total  = 16 * ksteps * 32 * 8;
    if (idx >= total) return;
    const int j    = idx & 7;
    const int lane = (idx >> 3) & 31;
    const int blk  = idx >> 8;           // ntile*ksteps + kstep
    const int nt   = blk / ksteps;
    const int kb   = (blk - nt * ksteps) << 5;
    const int k    = kb + ((j < 4) ? (8 * (lane >> 4) + 2 * j) : (16 + 8 * (lane >> 4) + 2 * (j - 4)));
    const int col  = nt * 16 + (lane & 15);
    out[idx] = pack2(B[(size_t)k * 256 + col], B[(size_t)(k + 1) * 256 + col]);
}

// ---------------- fp32 -> bf16 convert (for x) ----------------
__global__ void cvt_bf16_kernel(const float* __restrict__ src,
                                unsigned* __restrict__ dst, int npairs) {
    int i = blockIdx.x * blockDim.x + threadIdx.x;
    if (i < npairs) dst[i] = pack2(src[2 * i], src[2 * i + 1]);
}

// ---------------- WMMA inner tile ----------------
// Block = 256 threads (8 waves). Block output tile: 32 rows x 256 cols.
// Waves 0-3: row-tile 0; waves 4-7: row-tile 1. Wave covers 64 cols (4 n-tiles).
// A: bf16 [32 x K] contiguous at Ag (row-major, ld=K) -> LDS via async b128 copies.
// Bp: pre-packed bf16 B fragments.
template <int K>
__device__ __forceinline__ void gemm_tile(const unsigned short* __restrict__ Ag,
                                          const unsigned* __restrict__ Bp,
                                          unsigned* sAu, v8f acc[4]) {
    const int tid = threadIdx.x;

    // async-stage 32*K bf16 (= 32*K*2 bytes) into LDS, 16B per transfer
    {
        const unsigned lbase = lds_addr_of(sAu);
        const unsigned long long gbase = (unsigned long long)Ag;
        constexpr int CHUNKS = 32 * K * 2 / 16;
        for (int i = tid; i < CHUNKS; i += 256)
            async_copy_b128(lbase + i * 16, gbase + (unsigned long long)i * 16);
        wait_asynccnt0();
    }
    __syncthreads();

    const int lane    = tid & 31;
    const int wave    = tid >> 5;
    const int rowTile = wave >> 2;
    const int nt0     = (wave & 3) * 4;      // first of 4 n-tiles
    const int m       = lane & 15;
    const int half    = lane >> 4;
    const unsigned* sArow = sAu + (size_t)(rowTile * 16 + m) * (K / 2);

    acc[0] = (v8f){}; acc[1] = (v8f){}; acc[2] = (v8f){}; acc[3] = (v8f){};

    constexpr int KS = K / 32;
#pragma unroll
    for (int ks = 0; ks < KS; ++ks) {
        Frag a;
#pragma unroll
        for (int j = 0; j < 8; ++j) {
            const int pairIdx = (j < 4) ? (j + 4 * half) : (8 + (j - 4) + 4 * half);
            a.u[j] = sArow[ks * 16 + pairIdx];
        }
#pragma unroll
        for (int t = 0; t < 4; ++t) {
            Frag b;
            b.v = *(const v16bf*)(Bp + ((size_t)(nt0 + t) * KS + ks) * 256 + lane * 8);
            acc[t] = __builtin_amdgcn_wmma_f32_16x16x32_bf16(
                false, a.v, false, b.v, (short)0, acc[t], false, false);
        }
    }
}

// ---------------- Kernel 1: hb = bf16(relu(BN(x @ initW + b))) ----------------
__global__ void gemm_init_kernel(const unsigned short* __restrict__ xb,
                                 const unsigned* __restrict__ Bp,
                                 const float* __restrict__ bias,
                                 const float* __restrict__ bng,
                                 const float* __restrict__ bnb,
                                 const float* __restrict__ bnm,
                                 const float* __restrict__ bnv,
                                 unsigned short* __restrict__ hb) {
    __shared__ unsigned sAu[32 * F_IN / 2];
    v8f acc[4];
    gemm_tile<F_IN>(xb + (size_t)blockIdx.x * 32 * F_IN, Bp, sAu, acc);

    const int lane = threadIdx.x & 31;
    const int wave = threadIdx.x >> 5;
    const int row0 = blockIdx.x * 32 + (wave >> 2) * 16;
    const int n0   = (wave & 3) * 64;
    const int cn   = lane & 15;
    const int half = lane >> 4;
#pragma unroll
    for (int t = 0; t < 4; ++t) {
        const int col   = n0 + 16 * t + cn;
        const float scl = bng[col] * rsqrtf(bnv[col] + BN_EPS);
        const float sft = bnb[col] - bnm[col] * scl;
        const float bs  = bias[col];
#pragma unroll
        for (int j = 0; j < 8; ++j) {
            float y = (acc[t][j] + bs) * scl + sft;
            hb[(size_t)(row0 + j + 8 * half) * H0 + col] = bf16bits(fmaxf(y, 0.0f));
        }
    }
}

// ------- Kernel 2: y<4 -> xWb[y] = bf16(h @ W[y]);  y==4 -> agg = h @ root + bias -------
__global__ void gemm_rgcn_kernel(const unsigned short* __restrict__ hb,
                                 const unsigned* __restrict__ Bpack, // 5 packed mats
                                 const float* __restrict__ bias,
                                 unsigned short* __restrict__ xWb,   // [4][N][256] bf16
                                 float* __restrict__ agg) {          // [N][256] f32
    __shared__ unsigned sAu[32 * H0 / 2];
    const int y = blockIdx.y;
    const unsigned* Bp = Bpack + (size_t)y * (16 * (H0 / 32) * 32 * 8);

    v8f acc[4];
    gemm_tile<H0>(hb + (size_t)blockIdx.x * 32 * H0, Bp, sAu, acc);

    const int lane = threadIdx.x & 31;
    const int wave = threadIdx.x >> 5;
    const int row0 = blockIdx.x * 32 + (wave >> 2) * 16;
    const int n0   = (wave & 3) * 64;
    const int cn   = lane & 15;
    const int half = lane >> 4;

    if (y < 4) {
        unsigned short* outp = xWb + (size_t)y * NN * H1;
#pragma unroll
        for (int t = 0; t < 4; ++t) {
            const int col = n0 + 16 * t + cn;
#pragma unroll
            for (int j = 0; j < 8; ++j)
                outp[(size_t)(row0 + j + 8 * half) * H1 + col] = bf16bits(acc[t][j]);
        }
    } else {
#pragma unroll
        for (int t = 0; t < 4; ++t) {
            const int col  = n0 + 16 * t + cn;
            const float bs = bias[col];
#pragma unroll
            for (int j = 0; j < 8; ++j)
                agg[(size_t)(row0 + j + 8 * half) * H1 + col] = acc[t][j] + bs;
        }
    }
}

// ---------------- Kernel 3: edge scatter-add (bf16 msgs -> f32 atomics) ----------------
// blockDim = (32,8): 32 lanes x 8 bf16 cover 256 cols of one edge message.
__global__ void scatter_kernel(const int* __restrict__ ei,          // [2][E]
                               const int* __restrict__ etype,       // [E]
                               const unsigned short* __restrict__ xWb, // [4][N][256] bf16
                               float* __restrict__ agg) {           // [N][256] f32
    const int e = blockIdx.x * 8 + threadIdx.y;
    if (e >= EE) return;
    const int src = ei[e];
    const int dst = ei[EE + e];
    const int r   = etype[e];
    const uint4 mv =
        ((const uint4*)(xWb + ((size_t)r * NN + src) * H1))[threadIdx.x];
    float* a = agg + (size_t)dst * H1 + threadIdx.x * 8;
    atomicAdd(a + 0, bf2f((unsigned short)(mv.x & 0xffff)));
    atomicAdd(a + 1, bf2f((unsigned short)(mv.x >> 16)));
    atomicAdd(a + 2, bf2f((unsigned short)(mv.y & 0xffff)));
    atomicAdd(a + 3, bf2f((unsigned short)(mv.y >> 16)));
    atomicAdd(a + 4, bf2f((unsigned short)(mv.z & 0xffff)));
    atomicAdd(a + 5, bf2f((unsigned short)(mv.z >> 16)));
    atomicAdd(a + 6, bf2f((unsigned short)(mv.w & 0xffff)));
    atomicAdd(a + 7, bf2f((unsigned short)(mv.w >> 16)));
}

// ---------------- Kernel 4: hb = bf16(relu(agg)) ----------------
__global__ void relu_bf16_kernel(const float* __restrict__ agg,
                                 unsigned* __restrict__ hb2) {  // as u32 pairs
    const size_t n4 = (size_t)NN * H1 / 4;
    for (size_t i = blockIdx.x * (size_t)blockDim.x + threadIdx.x; i < n4;
         i += (size_t)gridDim.x * blockDim.x) {
        float4 v = ((const float4*)agg)[i];
        unsigned lo = pack2(fmaxf(v.x, 0.f), fmaxf(v.y, 0.f));
        unsigned hi = pack2(fmaxf(v.z, 0.f), fmaxf(v.w, 0.f));
        hb2[2 * i]     = lo;
        hb2[2 * i + 1] = hi;
    }
}

// ---------------- Kernel 5: zero output ----------------
__global__ void zero_kernel(float* __restrict__ p, int n) {
    int i = blockIdx.x * blockDim.x + threadIdx.x;
    if (i < n) p[i] = 0.0f;
}

// ---------------- Kernel 6: gated readout ----------------
__global__ void readout_kernel(const float* __restrict__ x,
                               const float* __restrict__ hom,
                               const unsigned short* __restrict__ hb,
                               const int*   __restrict__ batch,
                               const float* __restrict__ gW,  // [415]
                               const float* __restrict__ gb,
                               const float* __restrict__ gbg, const float* __restrict__ gbb,
                               const float* __restrict__ gbm, const float* __restrict__ gbv,
                               const float* __restrict__ tW,  // [256]
                               const float* __restrict__ tb,
                               const float* __restrict__ tbg, const float* __restrict__ tbb,
                               const float* __restrict__ tbm, const float* __restrict__ tbv,
                               float* __restrict__ out) {
    const int lane = threadIdx.x & 31;
    const int wave = threadIdx.x >> 5;
    const int node = blockIdx.x * 8 + wave;
    if (node >= NN) return;

    float s = 0.f, t = 0.f;
    const float* xn = x + (size_t)node * F_IN;
    const unsigned short* hn = hb + (size_t)node * H1;
    const float* mn = hom + (size_t)node * HOM;
    for (int i = lane; i < F_IN; i += 32) s += xn[i] * gW[i];
    for (int i = lane; i < H1; i += 32) {
        const float hv = bf2f(hn[i]);
        s += hv * gW[F_IN + i];
        t += hv * tW[i];
    }
    for (int i = lane; i < HOM; i += 32) s += mn[i] * gW[F_IN + H1 + i];

#pragma unroll
    for (int o = 16; o > 0; o >>= 1) {
        s += __shfl_xor(s, o, 32);
        t += __shfl_xor(t, o, 32);
    }
    if (lane == 0) {
        float gscl = gbg[0] * rsqrtf(gbv[0] + BN_EPS);
        float gate = (s + gb[0] - gbm[0]) * gscl + gbb[0];
        float tscl = tbg[0] * rsqrtf(tbv[0] + BN_EPS);
        float trans = (t + tb[0] - tbm[0]) * tscl + tbb[0];
        float p = trans / (1.0f + __expf(-gate));
        atomicAdd(&out[batch[node]], p);
    }
}

// ---------------- host-side launcher ----------------
extern "C" void kernel_launch(void* const* d_in, const int* in_sizes, int n_in,
                              void* d_out, int out_size, void* d_ws, size_t ws_size,
                              hipStream_t stream) {
    (void)in_sizes; (void)n_in; (void)out_size; (void)ws_size;

    const float* x    = (const float*)d_in[0];
    const float* hom  = (const float*)d_in[1];
    const int*   ei   = (const int*)d_in[2];
    const int*   et   = (const int*)d_in[3];
    const int*   fei  = (const int*)d_in[4];
    const int*   fet  = (const int*)d_in[5];
    const int*   bat  = (const int*)d_in[6];
    const float* iW   = (const float*)d_in[7];
    const float* ib   = (const float*)d_in[8];
    const float* ibng = (const float*)d_in[9];
    const float* ibnb = (const float*)d_in[10];
    const float* ibnm = (const float*)d_in[11];
    const float* ibnv = (const float*)d_in[12];
    const float* r0W  = (const float*)d_in[13];
    const float* r0r  = (const float*)d_in[14];
    const float* r0b  = (const float*)d_in[15];
    const float* r1W  = (const float*)d_in[16];
    const float* r1r  = (const float*)d_in[17];
    const float* r1b  = (const float*)d_in[18];
    const float* gW   = (const float*)d_in[19];
    const float* gb   = (const float*)d_in[20];
    const float* gbg  = (const float*)d_in[21];
    const float* gbb  = (const float*)d_in[22];
    const float* gbm  = (const float*)d_in[23];
    const float* gbv  = (const float*)d_in[24];
    const float* tW   = (const float*)d_in[25];
    const float* tb   = (const float*)d_in[26];
    const float* tbg  = (const float*)d_in[27];
    const float* tbb  = (const float*)d_in[28];
    const float* tbm  = (const float*)d_in[29];
    const float* tbv  = (const float*)d_in[30];
    float* out = (float*)d_out;

    // workspace layout
    float*          agg = (float*)d_ws;                                  // N*256 f32
    unsigned short* xWb = (unsigned short*)(agg + (size_t)NN * H1);      // 4*N*256 bf16
    unsigned short* hb  = xWb + (size_t)RR * NN * H1;                    // N*256 bf16
    unsigned short* xb  = hb + (size_t)NN * H1;                          // N*128 bf16
    unsigned*       bp  = (unsigned*)(xb + (size_t)NN * F_IN);

    const int PACK128 = 16 * (F_IN / 32) * 32 * 8;   // u32 per K=128 matrix
    const int PACK256 = 16 * (H0  / 32) * 32 * 8;    // u32 per K=256 matrix
    unsigned* bp_init = bp;                          // 1 matrix, K=128
    unsigned* bp_l0   = bp_init + PACK128;           // 5 matrices, K=256
    unsigned* bp_l1   = bp_l0 + 5 * (size_t)PACK256;

    // ---- pre-convert x to bf16, pre-pack weights into B-fragment layout ----
    const int XPAIRS = NN * F_IN / 2;
    cvt_bf16_kernel<<<(XPAIRS + 255) / 256, 256, 0, stream>>>(x, (unsigned*)xb, XPAIRS);

    pack_b_kernel<<<(PACK128 + 255) / 256, 256, 0, stream>>>(iW, bp_init, F_IN);
    for (int r = 0; r < 4; ++r)
        pack_b_kernel<<<(PACK256 + 255) / 256, 256, 0, stream>>>(
            r0W + (size_t)r * H0 * H1, bp_l0 + (size_t)r * PACK256, H0);
    pack_b_kernel<<<(PACK256 + 255) / 256, 256, 0, stream>>>(r0r, bp_l0 + 4 * (size_t)PACK256, H0);
    for (int r = 0; r < 4; ++r)
        pack_b_kernel<<<(PACK256 + 255) / 256, 256, 0, stream>>>(
            r1W + (size_t)r * H0 * H1, bp_l1 + (size_t)r * PACK256, H0);
    pack_b_kernel<<<(PACK256 + 255) / 256, 256, 0, stream>>>(r1r, bp_l1 + 4 * (size_t)PACK256, H0);

    const int MB = NN / 32;  // 1875 row-blocks

    // initial MLP
    gemm_init_kernel<<<MB, 256, 0, stream>>>(xb, bp_init, ib, ibng, ibnb, ibnm, ibnv, hb);

    // RGCN layer 0 (regular edges)
    gemm_rgcn_kernel<<<dim3(MB, 5), 256, 0, stream>>>(hb, bp_l0, r0b, xWb, agg);
    scatter_kernel<<<EE / 8, dim3(32, 8), 0, stream>>>(ei, et, xWb, agg);
    relu_bf16_kernel<<<2048, 256, 0, stream>>>(agg, (unsigned*)hb);

    // RGCN layer 1 (fully-adjacent edges)
    gemm_rgcn_kernel<<<dim3(MB, 5), 256, 0, stream>>>(hb, bp_l1, r1b, xWb, agg);
    scatter_kernel<<<EE / 8, dim3(32, 8), 0, stream>>>(fei, fet, xWb, agg);
    relu_bf16_kernel<<<2048, 256, 0, stream>>>(agg, (unsigned*)hb);

    // readout
    zero_kernel<<<(GG + 255) / 256, 256, 0, stream>>>(out, GG);
    readout_kernel<<<NN / 8, 256, 0, stream>>>(x, hom, hb, bat,
                                               gW, gb, gbg, gbb, gbm, gbv,
                                               tW, tb, tbg, tbb, tbm, tbv, out);
}
